// Linear3Bit_8254927143226
// MI455X (gfx1250) — compile-verified, matrix operand order checked by reading, not run on hardware
//
#include <hip/hip_runtime.h>

#define IN_F  4096
#define OUT_F 4096
#define GROUPS_PER_ROW (IN_F / 32)   // 128 groups of 32 along K per output row

typedef _Float16 v16h __attribute__((ext_vector_type(16)));
typedef _Float16 v8h  __attribute__((ext_vector_type(8)));
typedef float    v8f  __attribute__((ext_vector_type(8)));

// Load one 16-row f16 fragment slice from LDS following the CDNA5 16-bit
// A/B VGPR layout: lane<16 holds K {0..7, 16..23}, lane>=16 holds K {8..15, 24..31},
// i.e. two contiguous 16B runs per lane -> two ds_load_b128.
__device__ __forceinline__ v16h ldsFrag(const _Float16* rowptr, int koff) {
  union { v16h v; v8h h[2]; } u;
  u.h[0] = *(const v8h*)(rowptr + koff);        // K = koff .. koff+7
  u.h[1] = *(const v8h*)(rowptr + koff + 16);   // K = koff+16 .. koff+23
  return u.v;
}

// Decode one byte-triplet (8 3-bit codes) into q[0..7].
__device__ __forceinline__ void dec3(int b0, int b1, int b2, int* q) {
  q[0] =  b0 & 7;
  q[1] = (b0 >> 3) & 7;
  q[2] = ((b0 >> 6) & 3) | ((b1 & 1) << 2);
  q[3] = (b1 >> 1) & 7;
  q[4] = (b1 >> 4) & 7;
  q[5] = ((b1 >> 7) & 1) | ((b2 & 3) << 1);
  q[6] = (b2 >> 2) & 7;
  q[7] = (b2 >> 5) & 7;
}

__global__ __launch_bounds__(256)
void q3_linear_wmma(const float* __restrict__ x,
                    const int*   __restrict__ wq,
                    const float* __restrict__ wnorm,
                    const float* __restrict__ bias,
                    float*       __restrict__ out) {
  // Block tile: 128 (M) x 256 (N), K-step 32. Double-buffered, rows padded to
  // 40 halves (80B) to spread LDS bank accesses. Total 60 KB.
  __shared__ alignas(16) _Float16 Xs[2][128][40];
  __shared__ alignas(16) _Float16 Ws[2][256][40];

  const int tid = threadIdx.x;
  const int N0  = blockIdx.x * 256;   // output-feature tile base
  const int M0  = blockIdx.y * 128;   // token tile base

  // X staging: thread -> (row 0..127, half 0/1 of the 32-wide K slab)
  const int lr = tid >> 1;
  const int lh = tid & 1;

  // Wave tiling: 8 waves in 2 (M) x 4 (N); each wave owns 64x64 = 4x4 fragments.
  const int wave = tid >> 5;
  const int lane = tid & 31;
  const int wm   = wave & 1;          // 0..1 -> M
  const int wn   = wave >> 1;         // 0..3 -> N
  const int l16  = lane & 15;
  const int koff = (lane & 16) ? 8 : 0;

  // Running global pointers (strength-reduced; advance by constants per K-step).
  const float* xp = x + (size_t)(M0 + lr) * IN_F + lh * 16;
  const int*   wp = wq + (size_t)(N0 + tid) * (GROUPS_PER_ROW * 12);
  const float* sp = wnorm + (size_t)(N0 + tid) * GROUPS_PER_ROW;

  float4 xr[4];   // 16 floats of X
  int4   wb[3];   // 12 packed bytes = one full group (32 3-bit values)
  float  ws;

  auto load_global = [&]() {
    xr[0] = *(const float4*)(xp + 0);
    xr[1] = *(const float4*)(xp + 4);
    xr[2] = *(const float4*)(xp + 8);
    xr[3] = *(const float4*)(xp + 12);
    xp += 32;
    const int4* wp4 = (const int4*)wp;
    wb[0] = wp4[0]; wb[1] = wp4[1]; wb[2] = wp4[2];
    wp += 12;
    ws = *sp++;
  };

  auto store_lds = [&](int buf) {
    // ---- X: fp32 -> fp16, two b128 LDS stores ----
    v8h h0, h1;
    h0[0] = (_Float16)xr[0].x; h0[1] = (_Float16)xr[0].y;
    h0[2] = (_Float16)xr[0].z; h0[3] = (_Float16)xr[0].w;
    h0[4] = (_Float16)xr[1].x; h0[5] = (_Float16)xr[1].y;
    h0[6] = (_Float16)xr[1].z; h0[7] = (_Float16)xr[1].w;
    h1[0] = (_Float16)xr[2].x; h1[1] = (_Float16)xr[2].y;
    h1[2] = (_Float16)xr[2].z; h1[3] = (_Float16)xr[2].w;
    h1[4] = (_Float16)xr[3].x; h1[5] = (_Float16)xr[3].y;
    h1[6] = (_Float16)xr[3].z; h1[7] = (_Float16)xr[3].w;
    *(v8h*)&Xs[buf][lr][lh * 16 + 0] = h0;
    *(v8h*)&Xs[buf][lr][lh * 16 + 8] = h1;

    // ---- W: unpack 32 3-bit values (4 byte-triplets), scale, fp16 ----
    const int b[12] = {wb[0].x, wb[0].y, wb[0].z, wb[0].w,
                       wb[1].x, wb[1].y, wb[1].z, wb[1].w,
                       wb[2].x, wb[2].y, wb[2].z, wb[2].w};
    int q[32];
#pragma unroll
    for (int t = 0; t < 4; ++t)
      dec3(b[3 * t + 0], b[3 * t + 1], b[3 * t + 2], &q[8 * t]);
    const float sc  = ws * (2.0f / 7.0f);   // (q/7*2 - 1)*ws == q*sc - ws
    const float off = -ws;
#pragma unroll
    for (int t = 0; t < 4; ++t) {
      v8h w;
#pragma unroll
      for (int i = 0; i < 8; ++i) w[i] = (_Float16)fmaf((float)q[8 * t + i], sc, off);
      *(v8h*)&Ws[buf][tid][8 * t] = w;
    }
  };

  v8f acc[4][4] = {};   // f32 accumulators, 4 (M) x 4 (N) fragments per wave

  const int nk = IN_F / 32;   // 128 K-steps
  load_global();
  store_lds(0);

  for (int kb = 0; kb < nk; ++kb) {
    __syncthreads();                       // buf[kb&1] ready; buf[(kb+1)&1] free
    const int  cur  = kb & 1;
    const bool more = (kb + 1) < nk;
    if (more) load_global();               // issue next slab's loads early

    v16h b[4];
#pragma unroll
    for (int ni = 0; ni < 4; ++ni)
      b[ni] = ldsFrag(&Ws[cur][wn * 64 + ni * 16 + l16][0], koff);

#pragma unroll
    for (int mi = 0; mi < 4; ++mi) {
      const v16h a = ldsFrag(&Xs[cur][wm * 64 + mi * 16 + l16][0], koff);
#pragma unroll
      for (int ni = 0; ni < 4; ++ni)
        acc[mi][ni] = __builtin_amdgcn_wmma_f32_16x16x32_f16(
            false, a, false, b[ni], (short)0, acc[mi][ni], false, false);
    }

    if (more) store_lds(cur ^ 1);
  }

  // Epilogue: C/D layout — VGPR r: lane<16 -> M=r, lane>=16 -> M=r+8; N = lane%16.
  const int mhi = (lane & 16) ? 8 : 0;
#pragma unroll
  for (int mi = 0; mi < 4; ++mi) {
#pragma unroll
    for (int ni = 0; ni < 4; ++ni) {
      const int col = N0 + wn * 64 + ni * 16 + l16;
      const float bv = bias[col];
#pragma unroll
      for (int r = 0; r < 8; ++r) {
        const int row = M0 + wm * 64 + mi * 16 + mhi + r;
        out[(size_t)row * OUT_F + col] = acc[mi][ni][r] + bv;
      }
    }
  }
}

extern "C" void kernel_launch(void* const* d_in, const int* in_sizes, int n_in,
                              void* d_out, int out_size, void* d_ws, size_t ws_size,
                              hipStream_t stream) {
  const float* x     = (const float*)d_in[0];
  const int*   wq    = (const int*)d_in[1];
  const float* wnorm = (const float*)d_in[2];
  const float* bias  = (const float*)d_in[3];
  float*       out   = (float*)d_out;

  const int M = in_sizes[0] / IN_F;          // 2*2048 = 4096 tokens
  dim3 grid(OUT_F / 256, M / 128);           // 16 x 32 tiles
  q3_linear_wmma<<<grid, 256, 0, stream>>>(x, wq, wnorm, bias, out);
}
